// HLOCNetVLAD_11390253269644
// MI455X (gfx1250) — compile-verified, matrix operand order checked by reading, not run on hardware
//
#include <hip/hip_runtime.h>
#include <hip/hip_bf16.h>

#define EPS 1e-12f

typedef __attribute__((ext_vector_type(16))) __bf16 v16bf;
typedef __attribute__((ext_vector_type(8)))  float  v8f;

// Column (K-dim) index for element e (0..15) of a 16-bit WMMA A/B operand,
// per CDNA5 ISA 7.12.2: lanes 0-15 hold K=0..7 (elems 0-7) and K=16..23
// (elems 8-15); lanes 16-31 hold K=8..15 and K=24..31.
__device__ __forceinline__ int cmap(int e, int hi) {
    return e + ((e >= 8) ? 8 : 0) + (hi ? 8 : 0);
}

__global__ void k_zero(float* __restrict__ p, int n) {
    int i = blockIdx.x * blockDim.x + threadIdx.x;
    if (i < n) p[i] = 0.0f;
}

// ---------------------------------------------------------------------------
// Kernel 1: scores = conv_w @ feat (per b, 16-column tile of N), softmax over
// K=64 clusters, store assign (bf16, [B][N][K]) and accumulate asum[b][k].
// grid = B * (N/16), block = 128 (4 waves; wave w covers cluster rows w*16..).
// ---------------------------------------------------------------------------
__global__ void k_scores_softmax(const float* __restrict__ x,
                                 const float* __restrict__ conv_w,
                                 __hip_bfloat16* __restrict__ assign_bf,
                                 float* __restrict__ asum,
                                 int B, int D, int N, int K)
{
    const int NT = N / 16;
    const int nt = blockIdx.x % NT;
    const int b  = blockIdx.x / NT;
    const int n0 = nt * 16;
    const int t  = threadIdx.x;
    const int wv = t >> 5;
    const int l  = t & 31;
    const int hi = (l >= 16);
    const int m  = l & 15;
    const int k0 = wv * 16;

    __shared__ float featT[32][17];
    __shared__ float scoresS[64][17];

    const float* xb = x + (size_t)b * D * N;
    v8f c = {};

    for (int d0 = 0; d0 < D; d0 += 32) {
        for (int i = t; i < 32 * 16; i += 128) {
            int dd = i >> 4, nn = i & 15;
            featT[dd][nn] = xb[(size_t)(d0 + dd) * N + n0 + nn];
        }
        __syncthreads();
        v16bf a, bm;
#pragma unroll
        for (int e = 0; e < 16; ++e) {
            int col = cmap(e, hi);
            a[e]  = (__bf16)conv_w[(k0 + m) * D + d0 + col]; // A row = cluster k0+m
            bm[e] = (__bf16)featT[col][m];                   // B col = pixel n0+m
        }
        c = __builtin_amdgcn_wmma_f32_16x16x32_bf16(false, a, false, bm,
                                                    (short)0, c, false, false);
        __syncthreads();
    }

    // Spill scores tile to LDS (C layout: lane&15 = col, vgpr v -> row v+8*hi)
    {
        int n = l & 15;
#pragma unroll
        for (int v = 0; v < 8; ++v) {
            int mm = v + (hi ? 8 : 0);
            scoresS[k0 + mm][n] = c[v];
        }
    }
    __syncthreads();

    // Softmax over K per column n
    if (t < 16) {
        int n = t;
        float mx = -1e30f;
        for (int k = 0; k < K; ++k) mx = fmaxf(mx, scoresS[k][n]);
        float s = 0.0f;
        for (int k = 0; k < K; ++k) {
            float e = __expf(scoresS[k][n] - mx);
            scoresS[k][n] = e;
            s += e;
        }
        float inv = 1.0f / s;
        for (int k = 0; k < K; ++k) {
            float aa = scoresS[k][n] * inv;
            scoresS[k][n] = aa;
            assign_bf[((size_t)b * N + n0 + n) * K + k] = __float2bfloat16(aa);
        }
    }
    __syncthreads();

    // asum[b][k] += sum over this tile's 16 columns
    if (t < 64) {
        float p = 0.0f;
        for (int n = 0; n < 16; ++n) p += scoresS[t][n];
        atomicAdd(&asum[b * K + t], p);
    }
}

// ---------------------------------------------------------------------------
// Kernel 2: vlad[b][k][d] = sum_n feat[b][d][n]*assign[b][n][k] - asum*centers
// grid = B * (D/64) * (K/16), block = 128 (4 waves stacked over d).
// ---------------------------------------------------------------------------
__global__ void k_vlad(const float* __restrict__ x,
                       const __hip_bfloat16* __restrict__ assign_bf,
                       const float* __restrict__ asum,
                       const float* __restrict__ centers,
                       float* __restrict__ vlad,
                       int B, int D, int N, int K)
{
    const int kt  = blockIdx.x & 3;         // K/16 = 4
    const int dbk = (blockIdx.x >> 2) & 7;  // D/64 = 8
    const int b   = blockIdx.x >> 5;
    const int t = threadIdx.x, wv = t >> 5, l = t & 31;
    const int hi = (l >= 16), m = l & 15;
    const int d0 = dbk * 64, k0 = kt * 16;

    __shared__ float featT[64][33];
    __shared__ float asgT[32][17];

    const float* xb = x + (size_t)b * D * N;
    const __hip_bfloat16* ab = assign_bf + (size_t)b * N * K;
    v8f c = {};

    for (int nn0 = 0; nn0 < N; nn0 += 32) {
        for (int i = t; i < 64 * 32; i += 128) {
            int r = i >> 5, cc = i & 31;
            featT[r][cc] = xb[(size_t)(d0 + r) * N + nn0 + cc];
        }
        for (int i = t; i < 32 * 16; i += 128) {
            int r = i >> 4, cc = i & 15;
            asgT[r][cc] = __bfloat162float(ab[(size_t)(nn0 + r) * K + k0 + cc]);
        }
        __syncthreads();
        v16bf a, bm;
#pragma unroll
        for (int e = 0; e < 16; ++e) {
            int col = cmap(e, hi);
            a[e]  = (__bf16)featT[wv * 16 + m][col]; // A row = d0+wv*16+m
            bm[e] = (__bf16)asgT[col][m];            // B col = cluster k0+m
        }
        c = __builtin_amdgcn_wmma_f32_16x16x32_bf16(false, a, false, bm,
                                                    (short)0, c, false, false);
        __syncthreads();
    }

    int k = k0 + (l & 15);
    float as = asum[b * K + k];
#pragma unroll
    for (int v = 0; v < 8; ++v) {
        int mm = v + (hi ? 8 : 0);
        int d = d0 + wv * 16 + mm;
        vlad[((size_t)b * K + k) * D + d] = c[v] - as * centers[d * K + k];
    }
}

// ---------------------------------------------------------------------------
// Kernel 3: intra-normalize each (b,k) row over D, scatter into desc[b][d*K+k],
// accumulate descsq[b] (sum of squares after intra-norm) for global L2 norm.
// grid = B*K, block = 256.
// ---------------------------------------------------------------------------
__global__ void k_intranorm(const float* __restrict__ vlad,
                            float* __restrict__ desc,
                            float* __restrict__ descsq,
                            int B, int D, int K)
{
    int b = blockIdx.x / K, k = blockIdx.x % K;
    const float* row = vlad + ((size_t)b * K + k) * D;
    __shared__ float red[256];
    float ss = 0.0f;
    for (int d = threadIdx.x; d < D; d += 256) {
        float v = row[d];
        ss += v * v;
    }
    red[threadIdx.x] = ss;
    __syncthreads();
    for (int s = 128; s > 0; s >>= 1) {
        if (threadIdx.x < s) red[threadIdx.x] += red[threadIdx.x + s];
        __syncthreads();
    }
    float sum = red[0];
    float sc = 1.0f / fmaxf(sqrtf(sum), EPS);
    for (int d = threadIdx.x; d < D; d += 256)
        desc[(size_t)b * D * K + (size_t)d * K + k] = row[d] * sc;
    if (threadIdx.x == 0) atomicAdd(&descsq[b], sum * sc * sc);
}

// ---------------------------------------------------------------------------
// Kernel 4: whiten GEMM. M=16 (8 batches + zero pad), N = OUT in 16-tiles,
// Kdim = 32768 split into 8 chunks of 4096; partial results atomicAdd'ed.
// grid = 512, block = 128 -> 2048 waves, each owns one (ntile, jchunk).
// K-loop unrolled x4 so each iteration issues 4 WMMAs worth of b128 loads
// before the first s_wait_loadcnt; next 512B row chunk is prefetched
// (global_prefetch_b8). This stage streams the 536 MB weight matrix and is
// the HBM-bandwidth-bound critical path (~23 us at 23.3 TB/s).
// ---------------------------------------------------------------------------
__global__ void k_whiten(const float* __restrict__ desc,
                         const float* __restrict__ W,
                         float* __restrict__ raw,
                         int B, int DK, int OUT)
{
    const int t = threadIdx.x, wv = t >> 5, l = t & 31;
    const int hi = (l >= 16), m = l & 15;
    int gw    = blockIdx.x * 4 + wv;
    int ntile = gw & 255;   // OUT/16 = 256 tiles
    int jc    = gw >> 8;    // 0..7 K-chunks
    int o0    = ntile * 16;
    int jlen  = DK / 8;     // 4096
    int jbase = jc * jlen;
    const int o = o0 + m;
    const float* wrow = W + (size_t)o * DK + jbase;
    const float* arow = desc + (size_t)m * DK + jbase;
    const bool avalid = (m < B);

    v8f c = {};
    for (int j0 = 0; j0 < jlen; j0 += 128) {
        // Prefetch this lane's next 512B chunk of the weight row.
        __builtin_prefetch(wrow + j0 + 128, 0, 1);
#pragma unroll
        for (int u = 0; u < 4; ++u) {
            int jj = j0 + u * 32;
            v16bf a, bm;
#pragma unroll
            for (int e = 0; e < 16; ++e) {
                int col = cmap(e, hi);
                float av = avalid ? arow[jj + col] : 0.0f;
                a[e]  = (__bf16)av;
                bm[e] = (__bf16)wrow[jj + col];
            }
            c = __builtin_amdgcn_wmma_f32_16x16x32_bf16(false, a, false, bm,
                                                        (short)0, c, false, false);
        }
    }
#pragma unroll
    for (int v = 0; v < 8; ++v) {
        int mm = v + (hi ? 8 : 0);
        if (mm < B) atomicAdd(&raw[(size_t)mm * OUT + o0 + (l & 15)], c[v]);
    }
}

// ---------------------------------------------------------------------------
// Kernel 5: out[b][o] = L2norm( raw[b][o]*global_rnorm[b] + bias[o] )
// grid = B, block = 1024.
// ---------------------------------------------------------------------------
__global__ void k_finalize(const float* __restrict__ raw,
                           const float* __restrict__ bias,
                           const float* __restrict__ descsq,
                           float* __restrict__ out, int OUT)
{
    int b = blockIdx.x;
    __shared__ float red[1024];
    float rn = 1.0f / fmaxf(sqrtf(descsq[b]), EPS);
    float v[4];
    float ss = 0.0f;
#pragma unroll
    for (int i = 0; i < 4; ++i) {
        int o = threadIdx.x + i * 1024;
        v[i] = raw[(size_t)b * OUT + o] * rn + bias[o];
        ss += v[i] * v[i];
    }
    red[threadIdx.x] = ss;
    __syncthreads();
    for (int s = 512; s > 0; s >>= 1) {
        if (threadIdx.x < s) red[threadIdx.x] += red[threadIdx.x + s];
        __syncthreads();
    }
    float sc = 1.0f / fmaxf(sqrtf(red[0]), EPS);
#pragma unroll
    for (int i = 0; i < 4; ++i) {
        int o = threadIdx.x + i * 1024;
        out[(size_t)b * OUT + o] = v[i] * sc;
    }
}

extern "C" void kernel_launch(void* const* d_in, const int* in_sizes, int n_in,
                              void* d_out, int out_size, void* d_ws, size_t ws_size,
                              hipStream_t stream)
{
    (void)in_sizes; (void)n_in; (void)out_size; (void)ws_size;
    const float* x        = (const float*)d_in[0];
    const float* conv_w   = (const float*)d_in[1];
    const float* centers  = (const float*)d_in[2];
    const float* whiten_w = (const float*)d_in[3];
    const float* whiten_b = (const float*)d_in[4];
    float* out = (float*)d_out;

    const int B = 8, D = 512, N = 1600, K = 64, OUT = 4096, DK = D * K;

    // Workspace layout (all offsets 256B aligned):
    //  [0, 2048)                         asum      (B*K f32)
    //  [2048, 2304)                      descsq    (16 f32)
    //  [2304, 133376)                    raw       (B*OUT f32)
    //  [133376, 1771776)                 assign_bf (B*N*K bf16)
    //  [1771776, 2820352)                vlad      (B*K*D f32)
    //  [2820352, 3868928)                desc      (B*DK f32)
    char* ws = (char*)d_ws;
    float*          asum      = (float*)(ws);
    float*          descsq    = (float*)(ws + 2048);
    float*          raw       = (float*)(ws + 2304);
    __hip_bfloat16* assign_bf = (__hip_bfloat16*)(ws + 133376);
    float*          vlad      = (float*)(ws + 1771776);
    float*          desc      = (float*)(ws + 2820352);

    // Zero the accumulated region (asum|descsq|raw) every call.
    int nz = 133376 / 4;
    k_zero<<<(nz + 255) / 256, 256, 0, stream>>>((float*)ws, nz);

    k_scores_softmax<<<B * (N / 16), 128, 0, stream>>>(x, conv_w, assign_bf,
                                                       asum, B, D, N, K);
    k_vlad<<<B * (D / 64) * (K / 16), 128, 0, stream>>>(x, assign_bf, asum,
                                                        centers, vlad, B, D, N, K);
    k_intranorm<<<B * K, 256, 0, stream>>>(vlad, desc, descsq, B, D, K);
    k_whiten<<<512, 128, 0, stream>>>(desc, whiten_w, raw, B, DK, OUT);
    k_finalize<<<B, 1024, 0, stream>>>(raw, whiten_b, descsq, out, OUT);
}
